// ProtoMemory_41807211659725
// MI455X (gfx1250) — compile-verified, matrix operand order checked by reading, not run on hardware
//
#include <hip/hip_runtime.h>

// ---------------- problem constants ----------------
#define NIMG   8
#define CH     512
#define HWSZ   4096          // 64*64
#define FEAT   256
#define NUMK   512
#define HIDDEN 512
#define MTOT   (NIMG*HWSZ)   // 32768 spatial rows
#define TM     32            // M rows per workgroup
#define LAMBD  (1.0f/512.0f)

// LDS row strides chosen so row stride % 64 DWORDs != 0 -> conflict-free A fragments
#define LDB 520              // bf16 row stride for K=512 tiles (1040 B = 260 dw, %64 = 4)
#define LDQ 264              // bf16 row stride for K=256 tiles (528 B = 132 dw, %64 = 4)
#define LDF 512              // fp32 row stride (unpadded; P6a/P6b access it column-wise)

// packed bf16 weight layout offsets (in bf16 elements) inside d_ws
#define OFF_THETA  0
#define OFF_PROTOT (OFF_THETA  + FEAT*CH)      // 131072
#define OFF_PROTO  (OFF_PROTOT + NUMK*FEAT)    // 262144
#define OFF_FC1    (OFF_PROTO  + FEAT*NUMK)    // 393216
#define OFF_FC2    (OFF_FC1    + HIDDEN*NUMK)  // 655360
#define OFF_OW     (OFF_FC2    + NUMK*HIDDEN)  // 917504
#define WPACK_ELEMS (OFF_OW + CH*FEAT)         // 1048576 bf16 = 2 MB
#define WS_STATS_OFF ((size_t)WPACK_ELEMS * 2) // byte offset of BN stats

typedef __attribute__((ext_vector_type(16))) __bf16 bf16x16;
typedef __attribute__((ext_vector_type(8)))  __bf16 bf16x8;
typedef __attribute__((ext_vector_type(8)))  float  f32x8;

// ---------------- WMMA helpers ----------------
__device__ inline f32x8 wmma_bf16(bf16x16 a, bf16x16 b, f32x8 c) {
  // D = A(16x32 bf16) * B(32x16 bf16) + C(16x16 f32)
  return __builtin_amdgcn_wmma_f32_16x16x32_bf16(
      /*neg_a=*/false, a, /*neg_b=*/false, b,
      /*c_mod=*/(short)0, c, /*reuse_a=*/false, /*reuse_b=*/false);
}

// A operand from bf16 LDS tile, row-major [m][ldk].
// ISA 16-bit A layout: lane L<16 -> row L, K {0..7, 16..23}; lane L>=16 -> row L-16, K {8..15, 24..31}
__device__ inline bf16x16 load_A_bf16(const __bf16* S, int ldk, int m0, int k0, int lane) {
  int m   = m0 + (lane & 15);
  int off = (lane >> 4) << 3;
  const __bf16* p = S + m * ldk + k0 + off;
  bf16x16 a;
  *(bf16x8*)&a       = *(const bf16x8*)p;        // K = k0+off .. +7
  *((bf16x8*)&a + 1) = *(const bf16x8*)(p + 16); // K = k0+16+off .. +7
  return a;
}

// B operand from pre-swizzled global weights: tile (tn,tk) is 512 contiguous bf16,
// lane's 16 values are contiguous -> two global_load_b128 per lane, fully coalesced.
__device__ inline bf16x16 load_B(const __bf16* Wp, int tn, int tk, int ktiles, int lane) {
  const __bf16* p = Wp + ((size_t)(tn * ktiles + tk) << 9) + (lane << 4);
  bf16x16 b;
  *(bf16x8*)&b       = *(const bf16x8*)p;
  *((bf16x8*)&b + 1) = *(const bf16x8*)(p + 8);
  return b;
}

// ---------------- weight pack kernel ----------------
// Logical B matrix is [NN][KK] row-major; tr=true means element(n,k)=W[k*NN+n].
__device__ inline void pack_one(const float* __restrict__ W, __bf16* __restrict__ dst,
                                int NN, int KK, bool tr, int item) {
  int ktiles = KK >> 5;
  int t    = item >> 5;
  int lane = item & 31;
  int tn = t / ktiles, tk = t % ktiles;
  int n    = tn * 16 + (lane & 15);
  int kbase = tk * 32 + ((lane >> 4) << 3);
  __bf16* o = dst + ((size_t)t << 9) + (lane << 4);
#pragma unroll
  for (int j = 0; j < 8; ++j) {
    int k1 = kbase + j, k2 = kbase + 16 + j;
    float a = tr ? W[(size_t)k1 * NN + n] : W[(size_t)n * KK + k1];
    float b = tr ? W[(size_t)k2 * NN + n] : W[(size_t)n * KK + k2];
    o[j]     = (__bf16)a;
    o[8 + j] = (__bf16)b;
  }
}

__global__ __launch_bounds__(256) void pack_kernel(
    const float* __restrict__ theta, const float* __restrict__ proto,
    const float* __restrict__ fc1,   const float* __restrict__ fc2,
    const float* __restrict__ ow,    __bf16* __restrict__ wp,
    float* __restrict__ bn_sum, float* __restrict__ bn_sqs) {
  int tid = blockIdx.x * 256 + threadIdx.x;
  if (tid < 512)       bn_sum[tid] = 0.0f;
  else if (tid < 1024) bn_sqs[tid - 512] = 0.0f;

  if      (tid < 8192)  pack_one(theta, wp + OFF_THETA,  FEAT, CH,     false, tid);
  else if (tid < 16384) pack_one(proto, wp + OFF_PROTOT, NUMK, FEAT,   true,  tid - 8192);
  else if (tid < 24576) pack_one(proto, wp + OFF_PROTO,  FEAT, NUMK,   false, tid - 16384);
  else if (tid < 40960) pack_one(fc1,   wp + OFF_FC1,    HIDDEN, NUMK, false, tid - 24576);
  else if (tid < 57344) pack_one(fc2,   wp + OFF_FC2,    NUMK, HIDDEN, false, tid - 40960);
  else                  pack_one(ow,    wp + OFF_OW,     CH,   FEAT,   false, tid - 57344);
}

// ---------------- fused attention kernel ----------------
// LDS plan (113 KB total):
//   sR0 (64 KB, aliased): phase 1: x tile bf16 [TM][LDB]
//                         phase 3: logits fp32 [TM][LDF]
//                         phase 5: r=relu(fc1) bf16 [TM][LDB]
//                         phase 7: x/y fp32 [TM][LDF]
//   sB  (33 KB): attn bf16 [TM][LDB] (softmax out, fc2-residual updated in place)
//   sQ  (17 KB): q / read bf16 [TM][LDQ]
__global__ __launch_bounds__(256) void fused_kernel(
    const float* __restrict__ x, const __bf16* __restrict__ wp,
    const float* __restrict__ fc1_b, const float* __restrict__ fc2_b,
    const float* __restrict__ gamma_p,
    float* __restrict__ y, float* __restrict__ bn_sum, float* __restrict__ bn_sqs) {
  __shared__ __align__(16) char   sR0raw[TM * LDF * 4];  // 65536 B
  __shared__ __align__(16) __bf16 sB[TM * LDB];          // 33280 B
  __shared__ __align__(16) __bf16 sQ[TM * LDQ];          // 16896 B
  __bf16* sXb = (__bf16*)sR0raw;  // bf16 views (ld = LDB)
  float*  sF  = (float*)sR0raw;   // fp32 view  (ld = LDF)

  const int t     = threadIdx.x;
  const int lane  = t & 31;
  const int wv    = t >> 5;          // 8 wave32s
  const int mbase = blockIdx.x * TM;

  // ---- P0: stage x tile to LDS as bf16 (coalesced in hw) ----
  for (int idx = t; idx < TM * CH; idx += 256) {
    int c  = idx >> 5;
    int mm = idx & 31;
    int gm = mbase + mm;
    size_t gi = ((size_t)(gm >> 12) * CH + c) * HWSZ + (gm & 4095);
    sXb[mm * LDB + c] = (__bf16)x[gi];
  }
  __syncthreads();

  // ---- GEMM1: q = x @ theta^T  (N=256,K=512), 1/sqrt(FEAT) folded in ----
  for (int i = 0; i < 4; ++i) {
    int tile = wv + 8 * i;
    int tm = tile & 1, tn = tile >> 1;
    if (i < 3) __builtin_prefetch(wp + OFF_THETA + ((size_t)((tile + 8) >> 1) << 13), 0, 1);
    f32x8 acc = {};
    for (int tk = 0; tk < 16; ++tk)
      acc = wmma_bf16(load_A_bf16(sXb, LDB, tm * 16, tk * 32, lane),
                      load_B(wp + OFF_THETA, tn, tk, 16, lane), acc);
    int n  = tn * 16 + (lane & 15);
    int mr = tm * 16 + ((lane >> 4) << 3);
#pragma unroll
    for (int r = 0; r < 8; ++r)
      sQ[(mr + r) * LDQ + n] = (__bf16)(acc[r] * 0.0625f);
  }
  __syncthreads();

  // ---- GEMM2: logits = q @ proto  (N=512,K=256) -> fp32 LDS ----
  for (int i = 0; i < 8; ++i) {
    int tile = wv + 8 * i;
    int tm = tile & 1, tn = tile >> 1;
    if (i < 7) __builtin_prefetch(wp + OFF_PROTOT + ((size_t)((tile + 8) >> 1) << 12), 0, 1);
    f32x8 acc = {};
    for (int tk = 0; tk < 8; ++tk)
      acc = wmma_bf16(load_A_bf16(sQ, LDQ, tm * 16, tk * 32, lane),
                      load_B(wp + OFF_PROTOT, tn, tk, 8, lane), acc);
    int n  = tn * 16 + (lane & 15);
    int mr = tm * 16 + ((lane >> 4) << 3);
#pragma unroll
    for (int r = 0; r < 8; ++r) sF[(mr + r) * LDF + n] = acc[r];
  }
  __syncthreads();

  // ---- softmax + hard-shrink + L1 renorm; emit bf16 attn into sB ----
  for (int rr = 0; rr < 4; ++rr) {
    int mm = wv * 4 + rr;
    const float* row = sF + mm * LDF;
    float v[16];
    float mx = -3.4e38f;
#pragma unroll
    for (int j = 0; j < 16; ++j) { v[j] = row[lane + 32 * j]; mx = fmaxf(mx, v[j]); }
    for (int s = 16; s > 0; s >>= 1) mx = fmaxf(mx, __shfl_xor(mx, s, 32));
    float sum = 0.0f;
#pragma unroll
    for (int j = 0; j < 16; ++j) { v[j] = __expf(v[j] - mx); sum += v[j]; }
    for (int s = 16; s > 0; s >>= 1) sum += __shfl_xor(sum, s, 32);
    float inv = 1.0f / sum;
    float l1  = 0.0f;
#pragma unroll
    for (int j = 0; j < 16; ++j) {
      float p  = v[j] * inv;
      float d  = p - LAMBD;
      float sh = fmaxf(d, 0.0f) * p / (fabsf(d) + 1e-12f);
      v[j] = sh;
      l1  += sh;
    }
    for (int s = 16; s > 0; s >>= 1) l1 += __shfl_xor(l1, s, 32);
    float rn = 1.0f / (l1 + 1e-12f);
#pragma unroll
    for (int j = 0; j < 16; ++j)
      sB[mm * LDB + lane + 32 * j] = (__bf16)(v[j] * rn);
  }
  __syncthreads();

  // ---- GEMM3: r = relu(attn @ fc1^T + b1)  (N=512,K=512) -> sXb (logits dead) ----
  for (int i = 0; i < 8; ++i) {
    int tile = wv + 8 * i;
    int tm = tile & 1, tn = tile >> 1;
    if (i < 7) __builtin_prefetch(wp + OFF_FC1 + ((size_t)((tile + 8) >> 1) << 13), 0, 1);
    f32x8 acc = {};
    for (int tk = 0; tk < 16; ++tk)
      acc = wmma_bf16(load_A_bf16(sB, LDB, tm * 16, tk * 32, lane),
                      load_B(wp + OFF_FC1, tn, tk, 16, lane), acc);
    int n  = tn * 16 + (lane & 15);
    float bias = fc1_b[n];
    int mr = tm * 16 + ((lane >> 4) << 3);
#pragma unroll
    for (int r = 0; r < 8; ++r)
      sXb[(mr + r) * LDB + n] = (__bf16)fmaxf(acc[r] + bias, 0.0f);
  }
  __syncthreads();

  // ---- GEMM4: attn += r @ fc2^T + b2  (N=512,K=512); bf16 attn updated in place ----
  for (int i = 0; i < 8; ++i) {
    int tile = wv + 8 * i;
    int tm = tile & 1, tn = tile >> 1;
    if (i < 7) __builtin_prefetch(wp + OFF_FC2 + ((size_t)((tile + 8) >> 1) << 13), 0, 1);
    f32x8 acc = {};
    for (int tk = 0; tk < 16; ++tk)
      acc = wmma_bf16(load_A_bf16(sXb, LDB, tm * 16, tk * 32, lane),
                      load_B(wp + OFF_FC2, tn, tk, 16, lane), acc);
    int n  = tn * 16 + (lane & 15);
    float bias = fc2_b[n];
    int mr = tm * 16 + ((lane >> 4) << 3);
#pragma unroll
    for (int r = 0; r < 8; ++r) {
      int idx = (mr + r) * LDB + n;            // element owned by this lane only
      float a2 = (float)sB[idx] + acc[r] + bias;
      sB[idx] = (__bf16)a2;
    }
  }
  __syncthreads();

  // ---- GEMM5: read = attn @ proto^T  (N=256,K=512) -> sQ ----
  for (int i = 0; i < 4; ++i) {
    int tile = wv + 8 * i;
    int tm = tile & 1, tn = tile >> 1;
    if (i < 3) __builtin_prefetch(wp + OFF_PROTO + ((size_t)((tile + 8) >> 1) << 13), 0, 1);
    f32x8 acc = {};
    for (int tk = 0; tk < 16; ++tk)
      acc = wmma_bf16(load_A_bf16(sB, LDB, tm * 16, tk * 32, lane),
                      load_B(wp + OFF_PROTO, tn, tk, 16, lane), acc);
    int n  = tn * 16 + (lane & 15);
    int mr = tm * 16 + ((lane >> 4) << 3);
#pragma unroll
    for (int r = 0; r < 8; ++r) sQ[(mr + r) * LDQ + n] = (__bf16)acc[r];
  }
  __syncthreads();

  // ---- P6a: stage x fp32 into sF, coalesced (r tile in sR0 is dead) ----
  for (int idx = t; idx < TM * CH; idx += 256) {
    int c  = idx >> 5;
    int mm = idx & 31;
    int gm = mbase + mm;
    size_t gi = ((size_t)(gm >> 12) * CH + c) * HWSZ + (gm & 4095);
    sF[mm * LDF + c] = x[gi];
  }
  __syncthreads();

  // ---- GEMM6: y = gamma*(read @ o_w^T) + x  (N=512,K=256), in-place in sF + BN sums ----
  const float gamma = gamma_p[0];
  for (int i = 0; i < 8; ++i) {
    int tile = wv + 8 * i;
    int tm = tile & 1, tn = tile >> 1;
    if (i < 7) __builtin_prefetch(wp + OFF_OW + ((size_t)((tile + 8) >> 1) << 12), 0, 1);
    f32x8 acc = {};
    for (int tk = 0; tk < 8; ++tk)
      acc = wmma_bf16(load_A_bf16(sQ, LDQ, tm * 16, tk * 32, lane),
                      load_B(wp + OFF_OW, tn, tk, 8, lane), acc);
    int c  = tn * 16 + (lane & 15);      // channel: constant per lane
    int mr = tm * 16 + ((lane >> 4) << 3);
    float s1 = 0.0f, s2 = 0.0f;
#pragma unroll
    for (int r = 0; r < 8; ++r) {
      int fi = (mr + r) * LDF + c;
      float yv = gamma * acc[r] + sF[fi];
      sF[fi] = yv;
      s1 += yv;
      s2 += yv * yv;
    }
    s1 += __shfl_xor(s1, 16, 32);  // lanes L and L+16 share the same channel
    s2 += __shfl_xor(s2, 16, 32);
    if (lane < 16) {
      atomicAdd(&bn_sum[c], s1);
      atomicAdd(&bn_sqs[c], s2);
    }
  }
  __syncthreads();

  // ---- P6b: write y out, coalesced in hw ----
  for (int idx = t; idx < TM * CH; idx += 256) {
    int c  = idx >> 5;
    int mm = idx & 31;
    int gm = mbase + mm;
    size_t gi = ((size_t)(gm >> 12) * CH + c) * HWSZ + (gm & 4095);
    y[gi] = sF[mm * LDF + c];
  }
}

// ---------------- BatchNorm finalize ----------------
__global__ __launch_bounds__(256) void bn_stats_kernel(
    const float* __restrict__ bn_sum, const float* __restrict__ bn_sqs,
    const float* __restrict__ bn_w, const float* __restrict__ bn_b,
    float* __restrict__ scale, float* __restrict__ shift) {
  int c = blockIdx.x * 256 + threadIdx.x;
  if (c < CH) {
    const float invn = 1.0f / (float)MTOT;
    float mean = bn_sum[c] * invn;
    float var  = bn_sqs[c] * invn - mean * mean;
    float rstd = rsqrtf(var + 1e-5f);
    float sc   = bn_w[c] * rstd;
    scale[c] = sc;
    shift[c] = bn_b[c] - mean * sc;
  }
}

__global__ __launch_bounds__(256) void bn_apply_kernel(
    float* __restrict__ y, const float* __restrict__ scale,
    const float* __restrict__ shift) {
  size_t i = (size_t)blockIdx.x * 256 + threadIdx.x;
  int c = (int)((i >> 12) & 511);
  y[i] = y[i] * scale[c] + shift[c];
}

// ---------------- launcher ----------------
extern "C" void kernel_launch(void* const* d_in, const int* in_sizes, int n_in,
                              void* d_out, int out_size, void* d_ws, size_t ws_size,
                              hipStream_t stream) {
  const float* x     = (const float*)d_in[0];
  const float* theta = (const float*)d_in[1];
  const float* proto = (const float*)d_in[2];
  const float* fc1_w = (const float*)d_in[3];
  const float* fc1_b = (const float*)d_in[4];
  const float* fc2_w = (const float*)d_in[5];
  const float* fc2_b = (const float*)d_in[6];
  const float* ow    = (const float*)d_in[7];
  const float* gamma = (const float*)d_in[8];
  const float* bn_w  = (const float*)d_in[9];
  const float* bn_b  = (const float*)d_in[10];

  float* y  = (float*)d_out;
  char*  ws = (char*)d_ws;
  __bf16* wp     = (__bf16*)ws;
  float* bn_sum  = (float*)(ws + WS_STATS_OFF);
  float* bn_sqs  = bn_sum + 512;
  float* bn_scal = bn_sum + 1024;
  float* bn_shft = bn_sum + 1536;

  // 1) pack weights to WMMA-lane-ordered bf16 + zero BN accumulators
  pack_kernel<<<256, 256, 0, stream>>>(theta, proto, fc1_w, fc2_w, ow, wp, bn_sum, bn_sqs);
  // 2) fused attention pipeline: 1024 M-tiles of 32 rows
  fused_kernel<<<MTOT / TM, 256, 0, stream>>>(x, wp, fc1_b, fc2_b, gamma, y, bn_sum, bn_sqs);
  // 3) per-channel BN scale/shift
  bn_stats_kernel<<<2, 256, 0, stream>>>(bn_sum, bn_sqs, bn_w, bn_b, bn_scal, bn_shft);
  // 4) apply BN in place over d_out (8*512*4096 = 16.7M elements)
  bn_apply_kernel<<<65536, 256, 0, stream>>>(y, bn_scal, bn_shft);
}
